// CrossFrameInteractionGNN_74826920231630
// MI455X (gfx1250) — matrix operand order repaired and checked by reading, not compiled, not run on hardware
//
#include <hip/hip_runtime.h>
#include <cstdint>
#include <climits>

// ---------------------------------------------------------------------------
// CrossFrameInteractionGNN on MI455X (gfx1250, wave32, WMMA).
// GEMMs (edge encoder, edge-update MLP, message MLP, node-out) run on
// v_wmma_f32_16x16x32_bf16 with f32 accumulation. Gather/concat rows are
// staged into LDS as bf16 using CDNA5 async global->LDS copies
// (global_load_async_to_lds_b128, ASYNCcnt) in 16B chunks; weights are
// repacked once per launch into a fragment-ready layout so each B-frag is
// one contiguous 32B read per lane. Attention segment-softmax uses
// order-preserving int atomicMax + f32 atomicAdd; LayerNorm is wave-per-row
// with shfl_xor reductions.
// ---------------------------------------------------------------------------

#define N_NODES 50000
#define N_EDGES 200000
#define NODE_DIM 160
#define EDGE_DIM 64
#define RAW_EDGE_DIM 8
#define HID 256
#define ENC_HID 128

typedef __bf16 bf16_t;
typedef __attribute__((ext_vector_type(16))) bf16_t v16bf;
typedef __attribute__((ext_vector_type(8)))  bf16_t v8bf;
typedef __attribute__((ext_vector_type(8)))  float  v8f;

union Frag16 { v16bf v; v8bf h[2]; };

#define CDIV(a, b) (((a) + (b) - 1) / (b))

// ---------------------------------------------------------------------------
// Weight repack: f32 row-major [K,N] -> bf16 fragment-ready tiles.
// Tile (kt,nt) stores 32 lanes x 16 elems contiguous; lane = h*16+n,
// element p maps to B-matrix (k = kt*32 + (p>=8)*16 + h*8 + (p&7), col = nt*16 + n).
// Rows k >= K are zero padding (encoder K=328 padded to 352).
// ---------------------------------------------------------------------------
__global__ void repack_w_k(const float* __restrict__ W, bf16_t* __restrict__ out,
                           int K, int N, int KT) {
  int idx = blockIdx.x * 256 + threadIdx.x;
  int NT = N >> 4;
  int total = KT * NT * 512;
  if (idx >= total) return;
  int p    = idx & 15;
  int lane = (idx >> 4) & 31;
  int tile = idx >> 9;
  int nt = tile % NT;
  int kt = tile / NT;
  int k = kt * 32 + ((p & 8) << 1) + ((lane >> 4) << 3) + (p & 7);
  int n = nt * 16 + (lane & 15);
  float v = (k < K) ? W[(size_t)k * N + n] : 0.0f;
  out[idx] = (bf16_t)v;
}

// ---------------------------------------------------------------------------
// Fused (up to 3-way gather-concat) -> GEMM1(+bias,relu) -> [GEMM2(+bias)].
// ROWS rows per block, 2*ROWS threads (= ROWS/16 waves), one 16-row strip
// per wave. Staging: async global->LDS b128 per 16B chunk (all segment
// widths are multiples of 8 bf16). A frags: 16-bit A 16x32 layout (two 16B
// LDS chunks per lane). B frags: repacked global, one 32B read per lane.
// ---------------------------------------------------------------------------
template <int ROWS, int KPAD, int HIDN, int OUTN, bool TWO, bool RELU_OUT>
__global__ __launch_bounds__(ROWS * 2) void mlp_kernel(
    const bf16_t* __restrict__ s0, const int* __restrict__ i0, int w0,
    const bf16_t* __restrict__ s1, const int* __restrict__ i1, int w1,
    const bf16_t* __restrict__ s2, const int* __restrict__ i2, int w2,
    const bf16_t* __restrict__ W1p, const float* __restrict__ b1,
    const bf16_t* __restrict__ W2p, const float* __restrict__ b2,
    float* __restrict__ out, int nrows) {
  __shared__ bf16_t sA[ROWS * KPAD];
  __shared__ bf16_t sH[TWO ? ROWS * HIDN : 1];

  const int tid = threadIdx.x;
  const int blk = blockIdx.x;

  // ---- stage gathered+concatenated rows into LDS via async b128 copies ----
  {
    constexpr int CPR = KPAD / 8;  // 16B chunks per row
    for (int c = tid; c < ROWS * CPR; c += ROWS * 2) {
      int r = c / CPR;
      int k = (c - r * CPR) * 8;
      int g = blk * ROWS + r;
      if (g >= nrows) g = 0;  // clamp: out-of-range rows discarded at store
      uint32_t lds_off = (uint32_t)(uintptr_t)&sA[(size_t)r * KPAD + k];
      const bf16_t* src = nullptr;
      if (k < w0) {
        src = s0 + (size_t)(i0 ? i0[g] : g) * w0 + k;
      } else if (k < w0 + w1) {
        src = s1 + (size_t)(i1 ? i1[g] : g) * w1 + (k - w0);
      } else if (k < w0 + w1 + w2) {
        src = s2 + (size_t)(i2 ? i2[g] : g) * w2 + (k - w0 - w1);
      }
      if (src) {
        asm volatile("global_load_async_to_lds_b128 %0, %1, off"
                     :: "v"(lds_off), "v"(src)
                     : "memory");
      } else {
        v8bf z = {};  // K-padding (encoder 328 -> 352)
        *(v8bf*)&sA[(size_t)r * KPAD + k] = z;
      }
    }
    asm volatile("s_wait_asynccnt 0x0" ::: "memory");
  }
  __syncthreads();

  const int wv   = tid >> 5;
  const int lane = tid & 31;
  const int hh   = lane >> 4;   // half-wave: selects K sub-block for A frag
  const int mn   = lane & 15;   // A row within strip / C-D column

  // ---- layer 1: [ROWS,KPAD] x [KPAD,HIDN] ----
  {
    const bf16_t* arow = &sA[(size_t)(wv * 16 + mn) * KPAD];
    constexpr int NT1 = HIDN / 16;
    constexpr int KT1 = KPAD / 32;
    for (int nt = 0; nt < NT1; ++nt) {
      v8f acc = {};
      for (int kt = 0; kt < KT1; ++kt) {
        Frag16 a, b;
        a.h[0] = *(const v8bf*)(arow + kt * 32 + hh * 8);
        a.h[1] = *(const v8bf*)(arow + kt * 32 + 16 + hh * 8);
        b.v    = *(const v16bf*)(W1p + (size_t)((kt * NT1 + nt) * 32 + lane) * 16);
        acc = __builtin_amdgcn_wmma_f32_16x16x32_bf16(
            false, a.v, false, b.v, (short)0, acc, false, false);
      }
      float bias = b1[nt * 16 + mn];
      if constexpr (TWO) {
#pragma unroll
        for (int r = 0; r < 8; ++r) {
          float v = acc[r] + bias;
          v = v > 0.0f ? v : 0.0f;  // relu -> hidden, bf16 in LDS
          sH[(size_t)(wv * 16 + hh * 8 + r) * HIDN + nt * 16 + mn] = (bf16_t)v;
        }
      } else {
#pragma unroll
        for (int r = 0; r < 8; ++r) {
          int g = blk * ROWS + wv * 16 + hh * 8 + r;
          if (g < nrows) {
            float v = acc[r] + bias;
            if (RELU_OUT) v = v > 0.0f ? v : 0.0f;
            out[(size_t)g * HIDN + nt * 16 + mn] = v;
          }
        }
      }
    }
  }

  // ---- layer 2: [ROWS,HIDN] x [HIDN,OUTN] (same wave wrote its strip) ----
  if constexpr (TWO) {
    const bf16_t* hrow = &sH[(size_t)(wv * 16 + mn) * HIDN];
    constexpr int NT2 = OUTN / 16;
    constexpr int KT2 = HIDN / 32;
    for (int nt = 0; nt < NT2; ++nt) {
      v8f acc = {};
      for (int kt = 0; kt < KT2; ++kt) {
        Frag16 a, b;
        a.h[0] = *(const v8bf*)(hrow + kt * 32 + hh * 8);
        a.h[1] = *(const v8bf*)(hrow + kt * 32 + 16 + hh * 8);
        b.v    = *(const v16bf*)(W2p + (size_t)((kt * NT2 + nt) * 32 + lane) * 16);
        acc = __builtin_amdgcn_wmma_f32_16x16x32_bf16(
            false, a.v, false, b.v, (short)0, acc, false, false);
      }
      float bias = b2[nt * 16 + mn];
#pragma unroll
      for (int r = 0; r < 8; ++r) {
        int g = blk * ROWS + wv * 16 + hh * 8 + r;
        if (g < nrows) out[(size_t)g * OUTN + nt * 16 + mn] = acc[r] + bias;
      }
    }
  }
}

// ---------------------------------------------------------------------------
// Elementwise / reduction helper kernels
// ---------------------------------------------------------------------------
__global__ void fill_f32_k(float* p, float v, int n) {
  int i = blockIdx.x * 256 + threadIdx.x;
  if (i < n) p[i] = v;
}
__global__ void fill_i32_k(int* p, int v, int n) {
  int i = blockIdx.x * 256 + threadIdx.x;
  if (i < n) p[i] = v;
}
__global__ void cast_bf16_k(const float* __restrict__ s, bf16_t* __restrict__ d, int n) {
  int i = blockIdx.x * 256 + threadIdx.x;
  if (i < n) d[i] = (bf16_t)s[i];
}
// dst[row, 0:wa]=a[row], dst[row, wa:wa+wb]=b[row]  (f32 -> bf16)
__global__ void concat2_bf16_k(const float* __restrict__ a, int wa,
                               const float* __restrict__ b, int wb,
                               bf16_t* __restrict__ d, int total) {
  int i = blockIdx.x * 256 + threadIdx.x;
  if (i >= total) return;
  int W = wa + wb;
  int row = i / W;
  int j = i - row * W;
  float v = (j < wa) ? a[(size_t)row * wa + j] : b[(size_t)row * wb + (j - wa)];
  d[i] = (bf16_t)v;
}

// order-preserving int encoding of float for atomicMax
__device__ __forceinline__ int ordkey(float f) {
  int i = __float_as_int(f);
  return (i >= 0) ? i : (i ^ 0x7fffffff);
}
__device__ __forceinline__ float ordval(int k) {
  int i = (k >= 0) ? k : (k ^ 0x7fffffff);
  return __int_as_float(i);
}

// logit[e] = leaky_relu(dot(msg[e,:], att_v), 0.2); one wave per edge
__global__ __launch_bounds__(256) void logit_k(const float* __restrict__ msg,
                                               const float* __restrict__ attv,
                                               float* __restrict__ logit, int E) {
  int e = blockIdx.x * 8 + (threadIdx.x >> 5);
  int lane = threadIdx.x & 31;
  if (e >= E) return;
  float s = 0.0f;
#pragma unroll
  for (int k = lane; k < HID; k += 32) s += msg[(size_t)e * HID + k] * attv[k];
#pragma unroll
  for (int off = 16; off; off >>= 1) s += __shfl_xor(s, off, 32);
  if (lane == 0) logit[e] = (s > 0.0f) ? s : 0.2f * s;
}

__global__ void segmax_k(const float* __restrict__ logit, const int* __restrict__ row,
                         int* __restrict__ smax, int E) {
  int e = blockIdx.x * 256 + threadIdx.x;
  if (e < E) atomicMax(&smax[row[e]], ordkey(logit[e]));
}

__global__ void expden_k(const float* __restrict__ logit, const int* __restrict__ row,
                         const int* __restrict__ smax, float* __restrict__ ex,
                         float* __restrict__ den, int E) {
  int e = blockIdx.x * 256 + threadIdx.x;
  if (e >= E) return;
  int r = row[e];
  float v = expf(logit[e] - ordval(smax[r]));
  ex[e] = v;
  atomicAdd(&den[r], v);
}

// agg[row[e], :] += alpha_e * msg[e, :]; one wave per edge
__global__ __launch_bounds__(256) void agg_k(const float* __restrict__ msg,
                                             const float* __restrict__ ex,
                                             const float* __restrict__ den,
                                             const int* __restrict__ row,
                                             float* __restrict__ agg, int E) {
  int e = blockIdx.x * 8 + (threadIdx.x >> 5);
  int lane = threadIdx.x & 31;
  if (e >= E) return;
  int r = row[e];
  float alpha = ex[e] / (den[r] + 1e-16f);
#pragma unroll
  for (int k = lane; k < HID; k += 32)
    atomicAdd(&agg[(size_t)r * HID + k], alpha * msg[(size_t)e * HID + k]);
}

// out = LayerNorm(base + delta) * g + b; one wave per row, D <= 160
__global__ __launch_bounds__(256) void ln_k(const float* __restrict__ base,
                                            const float* __restrict__ delta,
                                            const float* __restrict__ g,
                                            const float* __restrict__ b,
                                            float* __restrict__ out, int n, int D) {
  int i = blockIdx.x * 8 + (threadIdx.x >> 5);
  int lane = threadIdx.x & 31;
  if (i >= n) return;
  float s = 0.0f, s2 = 0.0f;
  for (int k = lane; k < D; k += 32) {
    float v = base[(size_t)i * D + k] + delta[(size_t)i * D + k];
    s += v;
    s2 += v * v;
  }
#pragma unroll
  for (int off = 16; off; off >>= 1) {
    s += __shfl_xor(s, off, 32);
    s2 += __shfl_xor(s2, off, 32);
  }
  float mu = s / (float)D;
  float var = s2 / (float)D - mu * mu;
  float rs = rsqrtf(var + 1e-5f);
  for (int k = lane; k < D; k += 32) {
    float v = base[(size_t)i * D + k] + delta[(size_t)i * D + k];
    out[(size_t)i * D + k] = (v - mu) * rs * g[k] + b[k];
  }
}

// ---------------------------------------------------------------------------
// Host orchestration
// ---------------------------------------------------------------------------
extern "C" void kernel_launch(void* const* d_in, const int* in_sizes, int n_in,
                              void* d_out, int out_size, void* d_ws, size_t ws_size,
                              hipStream_t stream) {
  (void)in_sizes; (void)n_in; (void)out_size; (void)ws_size;

  const float* x      = (const float*)d_in[0];
  const float* eattr  = (const float*)d_in[1];
  const float* enc_W1 = (const float*)d_in[2];
  const float* enc_b1 = (const float*)d_in[3];
  const float* enc_W2 = (const float*)d_in[4];
  const float* enc_b2 = (const float*)d_in[5];
  const float* eW1    = (const float*)d_in[6];
  const float* eb1    = (const float*)d_in[7];
  const float* eW2    = (const float*)d_in[8];
  const float* eb2    = (const float*)d_in[9];
  const float* mW1    = (const float*)d_in[10];
  const float* mb1    = (const float*)d_in[11];
  const float* att_v  = (const float*)d_in[12];
  const float* nW1    = (const float*)d_in[13];
  const float* nb1    = (const float*)d_in[14];
  const float* ln_ng  = (const float*)d_in[15];
  const float* ln_nb  = (const float*)d_in[16];
  const float* ln_eg  = (const float*)d_in[17];
  const float* ln_eb  = (const float*)d_in[18];
  const int*   row    = (const int*)d_in[19];
  const int*   col    = (const int*)d_in[20];

  const int Nn = N_NODES, Ne = N_EDGES;

  // ---- workspace carving (256B aligned) ----
  char* ws = (char*)d_ws;
  size_t off = 0;
  auto alloc = [&](size_t bytes) -> void* {
    void* p = ws + off;
    off += (bytes + 255) & ~(size_t)255;
    return p;
  };
  bf16_t* encW1p = (bf16_t*)alloc((size_t)11 * 8 * 512 * 2);   // 352x128
  bf16_t* encW2p = (bf16_t*)alloc((size_t)4 * 4 * 512 * 2);    // 128x64
  bf16_t* eW1p   = (bf16_t*)alloc((size_t)24 * 16 * 512 * 2);  // 768x256
  bf16_t* eW2p   = (bf16_t*)alloc((size_t)8 * 4 * 512 * 2);    // 256x64
  bf16_t* mW1p   = (bf16_t*)alloc((size_t)12 * 16 * 512 * 2);  // 384x256
  bf16_t* nW1p   = (bf16_t*)alloc((size_t)8 * 10 * 512 * 2);   // 256x160
  bf16_t* x_bf   = (bf16_t*)alloc((size_t)Nn * NODE_DIM * 2);
  bf16_t* ea_bf  = (bf16_t*)alloc((size_t)Ne * RAW_EDGE_DIM * 2);
  bf16_t* nl_bf  = (bf16_t*)alloc((size_t)Nn * (2 * NODE_DIM) * 2);
  bf16_t* el_bf  = (bf16_t*)alloc((size_t)Ne * (2 * EDGE_DIM) * 2);
  bf16_t* ne_bf  = (bf16_t*)alloc((size_t)Ne * EDGE_DIM * 2);
  bf16_t* agg_bf = (bf16_t*)alloc((size_t)Nn * HID * 2);
  float* init_edge = (float*)alloc((size_t)Ne * EDGE_DIM * 4);
  float* edge_lat  = (float*)alloc((size_t)Ne * EDGE_DIM * 4);
  float* new_edge  = (float*)alloc((size_t)Ne * EDGE_DIM * 4);
  float* node_lat  = (float*)alloc((size_t)Nn * NODE_DIM * 4);
  float* new_node  = (float*)alloc((size_t)Nn * NODE_DIM * 4);
  float* msg       = (float*)alloc((size_t)Ne * HID * 4);
  float* logit     = (float*)alloc((size_t)Ne * 4);
  float* exv       = (float*)alloc((size_t)Ne * 4);
  int*   smax      = (int*)alloc((size_t)Nn * 4);
  float* den       = (float*)alloc((size_t)Nn * 4);
  float* agg       = (float*)alloc((size_t)Nn * HID * 4);

  auto repack = [&](const float* W, bf16_t* dst, int K, int N, int KT) {
    int total = KT * (N / 16) * 512;
    repack_w_k<<<CDIV(total, 256), 256, 0, stream>>>(W, dst, K, N, KT);
  };

  // ---- one-time prep ----
  repack(enc_W1, encW1p, 328, ENC_HID, 11);
  repack(enc_W2, encW2p, ENC_HID, EDGE_DIM, 4);
  repack(eW1, eW1p, 768, HID, 24);
  repack(eW2, eW2p, HID, EDGE_DIM, 8);
  repack(mW1, mW1p, 384, HID, 12);
  repack(nW1, nW1p, HID, NODE_DIM, 8);

  cast_bf16_k<<<CDIV(Nn * NODE_DIM, 256), 256, 0, stream>>>(x, x_bf, Nn * NODE_DIM);
  cast_bf16_k<<<CDIV(Ne * RAW_EDGE_DIM, 256), 256, 0, stream>>>(eattr, ea_bf, Ne * RAW_EDGE_DIM);

  // ---- edge encoder: [x[row], x[col], edge_attr] (328 -> pad 352) -> 128 -> 64 ----
  mlp_kernel<64, 352, ENC_HID, EDGE_DIM, true, false><<<CDIV(Ne, 64), 128, 0, stream>>>(
      x_bf, row, NODE_DIM, x_bf, col, NODE_DIM, ea_bf, nullptr, RAW_EDGE_DIM,
      encW1p, enc_b1, encW2p, enc_b2, init_edge, Ne);

  // node_lat = x ; edge_lat = init_edge
  hipMemcpyAsync(node_lat, x, (size_t)Nn * NODE_DIM * 4, hipMemcpyDeviceToDevice, stream);
  hipMemcpyAsync(edge_lat, init_edge, (size_t)Ne * EDGE_DIM * 4, hipMemcpyDeviceToDevice, stream);

  for (int it = 0; it < 3; ++it) {
    // nl = [x, node_lat] (bf16), el = [init_edge, edge_lat] (bf16)
    concat2_bf16_k<<<CDIV(Nn * 2 * NODE_DIM, 256), 256, 0, stream>>>(
        x, NODE_DIM, node_lat, NODE_DIM, nl_bf, Nn * 2 * NODE_DIM);
    concat2_bf16_k<<<CDIV(Ne * 2 * EDGE_DIM, 256), 256, 0, stream>>>(
        init_edge, EDGE_DIM, edge_lat, EDGE_DIM, el_bf, Ne * 2 * EDGE_DIM);

    // edge update: [nl[row], nl[col], el] (768) -> 256 -> 64
    mlp_kernel<32, 768, HID, EDGE_DIM, true, false><<<CDIV(Ne, 32), 64, 0, stream>>>(
        nl_bf, row, 2 * NODE_DIM, nl_bf, col, 2 * NODE_DIM, el_bf, nullptr, 2 * EDGE_DIM,
        eW1p, eb1, eW2p, eb2, new_edge, Ne);

    cast_bf16_k<<<CDIV(Ne * EDGE_DIM, 256), 256, 0, stream>>>(new_edge, ne_bf, Ne * EDGE_DIM);

    // message: relu([nl[col], new_edge] (384) @ mW1 + mb1) -> msg [E,256]
    mlp_kernel<64, 384, HID, 16, false, true><<<CDIV(Ne, 64), 128, 0, stream>>>(
        nl_bf, col, 2 * NODE_DIM, ne_bf, nullptr, EDGE_DIM, nullptr, nullptr, 0,
        mW1p, mb1, nullptr, nullptr, msg, Ne);

    // attention: logit, segment max, exp/den, weighted scatter
    logit_k<<<CDIV(Ne, 8), 256, 0, stream>>>(msg, att_v, logit, Ne);
    fill_i32_k<<<CDIV(Nn, 256), 256, 0, stream>>>(smax, INT_MIN, Nn);
    segmax_k<<<CDIV(Ne, 256), 256, 0, stream>>>(logit, row, smax, Ne);
    fill_f32_k<<<CDIV(Nn, 256), 256, 0, stream>>>(den, 0.0f, Nn);
    expden_k<<<CDIV(Ne, 256), 256, 0, stream>>>(logit, row, smax, exv, den, Ne);
    fill_f32_k<<<CDIV(Nn * HID, 256), 256, 0, stream>>>(agg, 0.0f, Nn * HID);
    agg_k<<<CDIV(Ne, 8), 256, 0, stream>>>(msg, exv, den, row, agg, Ne);

    // node out: agg [N,256] @ nW1 + nb1 -> new_node [N,160]
    cast_bf16_k<<<CDIV(Nn * HID, 256), 256, 0, stream>>>(agg, agg_bf, Nn * HID);
    mlp_kernel<64, 256, NODE_DIM, 16, false, false><<<CDIV(Nn, 64), 128, 0, stream>>>(
        agg_bf, nullptr, HID, nullptr, nullptr, 0, nullptr, nullptr, 0,
        nW1p, nb1, nullptr, nullptr, new_node, Nn);

    // residual + shared LayerNorms
    ln_k<<<CDIV(Nn, 8), 256, 0, stream>>>(x, new_node, ln_ng, ln_nb, node_lat, Nn, NODE_DIM);
    ln_k<<<CDIV(Ne, 8), 256, 0, stream>>>(init_edge, new_edge, ln_eg, ln_eb, edge_lat, Ne, EDGE_DIM);
  }

  // pack tuple output: node_lat [N,160] then edge_lat [E,64]
  hipMemcpyAsync(d_out, node_lat, (size_t)Nn * NODE_DIM * 4, hipMemcpyDeviceToDevice, stream);
  hipMemcpyAsync((float*)d_out + (size_t)Nn * NODE_DIM, edge_lat,
                 (size_t)Ne * EDGE_DIM * 4, hipMemcpyDeviceToDevice, stream);
}
